// Repro_11879879542246
// MI455X (gfx1250) — matrix operand order repaired and checked by reading, not compiled, not run on hardware
//
#include <hip/hip_runtime.h>
#include <hip/hip_bf16.h>

typedef __attribute__((ext_vector_type(16))) _Float16 v16h;
typedef __attribute__((ext_vector_type(8)))  float    v8f;

#define D_MODEL  256
#define N_HEAD   4
#define HEAD_DIM 64
#define SEQ      2048
#define BATCH    4
#define M_TOT    (BATCH * SEQ)     // 8192
#define E_TOT    (3 * D_MODEL)     // 768

// ---------------------------------------------------------------------------
// Kernel 1: fused QKV projection.  qkv[m,e] = sum_d x[m,d] * W[e,d] + b[e]
// One wave per 16x16 tile; K-loop of 8 x v_wmma_f32_16x16x32_f16.
// Q is pre-scaled by 1/sqrt(64).  Q/K stored f16 [B,H,S,64]; V stored
// TRANSPOSED f16 [B,H,64,S] so attention's P·V B-fragments are contiguous.
// ---------------------------------------------------------------------------
__global__ __launch_bounds__(256) void qkv_proj_kernel(
    const float* __restrict__ x, const float* __restrict__ W,
    const float* __restrict__ bias,
    _Float16* __restrict__ qh, _Float16* __restrict__ kh, _Float16* __restrict__ vT)
{
    const int wave = threadIdx.x >> 5;
    const int lane = threadIdx.x & 31;
    const int l16  = lane & 15;
    const int hlf  = lane >> 4;

    const int tile  = blockIdx.x * 8 + wave;       // 24576 tiles total
    const int tileM = tile / (E_TOT / 16);
    const int tileN = tile % (E_TOT / 16);

    const int row = tileM * 16 + l16;              // global row (B*S)
    const int col = tileN * 16 + l16;              // global e in [0,768)
    const float* xrow = x + (size_t)row * D_MODEL;
    const float* wrow = W + (size_t)col * D_MODEL;

    v8f c = {};
    for (int kk = 0; kk < D_MODEL; kk += 32) {
        v16h a, b;
        const int ka = kk + hlf * 16;   // A layout: K = ka+i, contiguous 16
        const int kb = kk + hlf * 16;   // B layout: K = kb+i, contiguous 16
#pragma unroll
        for (int i = 0; i < 8; ++i) {
            a[i]     = (_Float16)xrow[ka + i];
            a[i + 8] = (_Float16)xrow[ka + 8 + i];
        }
#pragma unroll
        for (int i = 0; i < 16; ++i) b[i] = (_Float16)wrow[kb + i];
        c = __builtin_amdgcn_wmma_f32_16x16x32_f16(false, a, false, b,
                                                   (short)0, c, false, false);
    }

    const float bi    = bias[col];
    const int   which = col >> 8;        // 0 = Q, 1 = K, 2 = V
    const int   em    = col & 255;
    const int   h     = em >> 6;
    const int   dh    = em & 63;
    const float scl   = (which == 0) ? 0.125f : 1.0f;   // fold 1/sqrt(64) into Q

#pragma unroll
    for (int r = 0; r < 8; ++r) {
        const int m  = tileM * 16 + hlf * 8 + r;   // C layout: M = r + half*8
        const int bb = m >> 11;                    // m / SEQ
        const int ss = m & (SEQ - 1);
        const _Float16 val = (_Float16)((c[r] + bi) * scl);
        const size_t bhh = (size_t)bb * N_HEAD + h;
        if (which == 2) {
            vT[(bhh * HEAD_DIM + dh) * SEQ + ss] = val;          // [B,H,64,S]
        } else {
            _Float16* dst = (which == 0) ? qh : kh;
            dst[(bhh * SEQ + ss) * HEAD_DIM + dh] = val;         // [B,H,S,64]
        }
    }
}

// ---------------------------------------------------------------------------
// Kernel 2: flash attention.  One wave per 32-query tile of one (b,h).
// Per 32-key iteration: 8 WMMAs (scores, 2 M-tiles x 2 N-tiles x 2 k-steps),
// online softmax, LDS transpose of P, 8 WMMAs (P·V with V fragments reused
// across both M-tiles).  Output fp32 [B,H,S,Dh].
// ---------------------------------------------------------------------------
__global__ __launch_bounds__(128) void attn_kernel(
    const _Float16* __restrict__ qh, const _Float16* __restrict__ kh,
    const _Float16* __restrict__ vT, float* __restrict__ out)
{
    __shared__ _Float16 pstage[4][32 * 32];   // per-wave P transpose slab

    const int wave = threadIdx.x >> 5;
    const int lane = threadIdx.x & 31;
    const int l16  = lane & 15;
    const int hlf  = lane >> 4;

    const int qt    = blockIdx.x * 4 + wave;   // 0..1023
    const int bh    = qt >> 6;                 // b*H + h   (S/32 = 64)
    const int qtile = qt & 63;                 // 32-row query tile

    const _Float16* qp = qh + (size_t)bh * SEQ * HEAD_DIM;
    const _Float16* kp = kh + (size_t)bh * SEQ * HEAD_DIM;
    const _Float16* vp = vT + (size_t)bh * HEAD_DIM * SEQ;

    // Q: 2 M-tiles x 2 k-steps of A-layout 16x32 fragments
    v16h qa[2][2];
#pragma unroll
    for (int mt = 0; mt < 2; ++mt) {
        const _Float16* qr =
            qp + (size_t)(qtile * 32 + mt * 16 + l16) * HEAD_DIM;
#pragma unroll
        for (int t = 0; t < 2; ++t) {
            const int dbase = t * 32 + hlf * 16;
#pragma unroll
            for (int i = 0; i < 8; ++i) {
                qa[mt][t][i]     = qr[dbase + i];
                qa[mt][t][i + 8] = qr[dbase + 8 + i];
            }
        }
    }

    v8f o[2][4] = {};
    float mrow[2][8], lrow[2][8];
#pragma unroll
    for (int mt = 0; mt < 2; ++mt)
#pragma unroll
        for (int r = 0; r < 8; ++r) { mrow[mt][r] = -1e30f; lrow[mt][r] = 0.0f; }

    _Float16* st = pstage[wave];

    for (int kt = 0; kt < SEQ; kt += 32) {
        // ---- K fragments (B layout, contiguous 16 halfs each) ----
        v16h kb0[2], kb1[2];
#pragma unroll
        for (int t = 0; t < 2; ++t) {
            const int dbase = t * 32 + hlf * 16;
            const _Float16* k0 = kp + (size_t)(kt + l16) * HEAD_DIM + dbase;
            const _Float16* k1 = k0 + 16 * HEAD_DIM;
#pragma unroll
            for (int i = 0; i < 16; ++i) { kb0[t][i] = k0[i]; kb1[t][i] = k1[i]; }
        }

        // ---- scores: sc[mt][nt], 8 WMMAs ----
        v8f sc[2][2] = {};
#pragma unroll
        for (int mt = 0; mt < 2; ++mt)
#pragma unroll
            for (int t = 0; t < 2; ++t) {
                sc[mt][0] = __builtin_amdgcn_wmma_f32_16x16x32_f16(
                    false, qa[mt][t], false, kb0[t], (short)0, sc[mt][0], false, false);
                sc[mt][1] = __builtin_amdgcn_wmma_f32_16x16x32_f16(
                    false, qa[mt][t], false, kb1[t], (short)0, sc[mt][1], false, false);
            }

        // ---- online softmax per M-tile ----
        float alpha[2][8];
#pragma unroll
        for (int mt = 0; mt < 2; ++mt) {
#pragma unroll
            for (int r = 0; r < 8; ++r) {
                float mx = fmaxf(sc[mt][0][r], sc[mt][1][r]);
#pragma unroll
                for (int off = 1; off < 16; off <<= 1)
                    mx = fmaxf(mx, __shfl_xor(mx, off, 32));
                const float mnew = fmaxf(mrow[mt][r], mx);
                alpha[mt][r] = __expf(mrow[mt][r] - mnew);
                const float p0 = __expf(sc[mt][0][r] - mnew);
                const float p1 = __expf(sc[mt][1][r] - mnew);
                sc[mt][0][r] = p0; sc[mt][1][r] = p1;
                float rs = p0 + p1;
#pragma unroll
                for (int off = 1; off < 16; off <<= 1)
                    rs += __shfl_xor(rs, off, 32);
                lrow[mt][r] = lrow[mt][r] * alpha[mt][r] + rs;
                mrow[mt][r] = mnew;
            }
#pragma unroll
            for (int nt = 0; nt < 4; ++nt)
#pragma unroll
                for (int r = 0; r < 8; ++r) o[mt][nt][r] *= alpha[mt][r];
        }

        // ---- P: C-layout -> LDS [32 rows][32 keys] -> reload in A layout ----
#pragma unroll
        for (int mt = 0; mt < 2; ++mt)
#pragma unroll
            for (int r = 0; r < 8; ++r) {
                const int prow = mt * 16 + hlf * 8 + r;
                st[prow * 32 + l16]      = (_Float16)sc[mt][0][r];
                st[prow * 32 + 16 + l16] = (_Float16)sc[mt][1][r];
            }
        v16h pa[2];
#pragma unroll
        for (int mt = 0; mt < 2; ++mt) {
            const int kbs = hlf * 16;
            const _Float16* pr = st + (mt * 16 + l16) * 32;
#pragma unroll
            for (int i = 0; i < 8; ++i) {
                pa[mt][i]     = pr[kbs + i];
                pa[mt][i + 8] = pr[kbs + 8 + i];
            }
        }

        // ---- P(16x32) x V(32x64): V fragments contiguous (transposed V),
        //      reused across both M-tiles; 8 WMMAs ----
#pragma unroll
        for (int nt = 0; nt < 4; ++nt) {
            v16h vb;
            const _Float16* v0 =
                vp + (size_t)(nt * 16 + l16) * SEQ + kt + hlf * 16;
#pragma unroll
            for (int i = 0; i < 16; ++i) vb[i] = v0[i];
#pragma unroll
            for (int mt = 0; mt < 2; ++mt)
                o[mt][nt] = __builtin_amdgcn_wmma_f32_16x16x32_f16(
                    false, pa[mt], false, vb, (short)0, o[mt][nt], false, false);
        }
    }

    // ---- epilogue: normalize and store fp32 [B,H,S,Dh] ----
#pragma unroll
    for (int mt = 0; mt < 2; ++mt)
#pragma unroll
        for (int nt = 0; nt < 4; ++nt)
#pragma unroll
            for (int r = 0; r < 8; ++r) {
                const int srow = qtile * 32 + mt * 16 + hlf * 8 + r;
                out[((size_t)bh * SEQ + srow) * HEAD_DIM + nt * 16 + l16] =
                    o[mt][nt][r] / lrow[mt][r];
            }
}

// ---------------------------------------------------------------------------
extern "C" void kernel_launch(void* const* d_in, const int* in_sizes, int n_in,
                              void* d_out, int out_size, void* d_ws, size_t ws_size,
                              hipStream_t stream) {
    const float* x = (const float*)d_in[0];
    const float* W = (const float*)d_in[1];
    const float* b = (const float*)d_in[2];
    float* out = (float*)d_out;

    const size_t per_tensor = (size_t)BATCH * N_HEAD * SEQ * HEAD_DIM;  // 2M elems
    _Float16* qh = (_Float16*)d_ws;
    _Float16* kh = qh + per_tensor;
    _Float16* vT = kh + per_tensor;

    // 24576 tiles / 8 waves per 256-thread block
    qkv_proj_kernel<<<(M_TOT / 16) * (E_TOT / 16) / 8, 256, 0, stream>>>(
        x, W, b, qh, kh, vT);

    // 1024 query tiles (32 rows each) / 4 waves per 128-thread block
    attn_kernel<<<(BATCH * N_HEAD * (SEQ / 32)) / 4, 128, 0, stream>>>(
        qh, kh, vT, out);
}
